// MultiHeadXCITAttention_17815524344165
// MI455X (gfx1250) — matrix operand order repaired and checked by reading, not compiled, not run on hardware
//
#include <hip/hip_runtime.h>
#include <stdint.h>

// ---------------------------------------------------------------------------
// XCiT cross-covariance attention for MI455X (gfx1250).
// wave32 + v_wmma_f32_16x16x32_f16 + TDM tensor_load_to_lds double buffering.
// ---------------------------------------------------------------------------

typedef _Float16 half_t;
typedef __attribute__((ext_vector_type(16))) _Float16 v16h;
typedef __attribute__((ext_vector_type(8)))  _Float16 v8h;
typedef __attribute__((ext_vector_type(8)))  float    v8f;
typedef __attribute__((ext_vector_type(4)))  unsigned int u32x4;
typedef __attribute__((ext_vector_type(4)))  int i32x4;
typedef __attribute__((ext_vector_type(8)))  int i32x8;

constexpr int kB = 16;
constexpr int kN = 4096;
constexpr int kD = 768;
constexpr int kH = 16;
constexpr int kE = 48;
constexpr int kTOK = kB * kN;     // 65536 tokens
constexpr int LDSTRIDE = 40;      // 32 K-elems + 8 pad (80B rows, 16B aligned)

#if defined(__HIP_DEVICE_COMPILE__) && __has_builtin(__builtin_amdgcn_tensor_load_to_lds)
#define HAVE_TDM 1
#else
#define HAVE_TDM 0
#endif

__device__ __forceinline__ v8f wmma16(v16h a, v16h b, v8f c) {
  return __builtin_amdgcn_wmma_f32_16x16x32_f16(false, a, false, b, (short)0, c,
                                                false, false);
}

__device__ __forceinline__ v16h pack16(v8h lo, v8h hi) {
  v16h r;
#pragma unroll
  for (int i = 0; i < 8; ++i) { r[i] = lo[i]; r[8 + i] = hi[i]; }
  return r;
}

// A fragment (16x32 f16): lane L holds row (L&15); K pairs at kb..kb+7 and
// kb+16..kb+23 with kb = (L<16)?0:8.
__device__ __forceinline__ v16h fragA(const half_t* base, int stride, int lane) {
  const half_t* p = base + (size_t)(lane & 15) * stride + ((lane < 16) ? 0 : 8);
  v8h lo = *(const v8h*)p;
  v8h hi = *(const v8h*)(p + 16);
  return pack16(lo, hi);
}

// B fragment (32x16 f16): lane L holds column (L&15); K at kb..kb+7 and
// kb+16..kb+23 with kb = (L<16)?0:8.
__device__ __forceinline__ v16h fragB(const half_t* base, int stride, int lane) {
  const half_t* p = base + (size_t)(lane & 15) * stride + ((lane < 16) ? 0 : 8);
  v8h lo = *(const v8h*)p;
  v8h hi = *(const v8h*)(p + 16);
  return pack16(lo, hi);
}

#if HAVE_TDM
// Issue one TDM 2D tile load: 128 rows x 64 bytes, global row stride 1536 B,
// into LDS with 16 B of padding appended per 64 B row (-> 80 B LDS rows,
// matching LDSTRIDE = 40 halfs). data_size = 4 B units.
__device__ __forceinline__ void tdm_load_tile(const half_t* g, unsigned lds_off) {
  const unsigned long long ga = (unsigned long long)(size_t)g;
  const unsigned TD0 = 0x100000u;  // huge tensor dims: tile is always in-bounds
  const unsigned TD1 = 0x100000u;
  u32x4 g0;
  g0[0] = 1u;                                   // count=1, user mode, no gather
  g0[1] = lds_off;                              // lds_addr (bytes)
  g0[2] = (unsigned)(ga & 0xFFFFFFFFu);         // global_addr[31:0]
  g0[3] = (unsigned)((ga >> 32) & 0x01FFFFFFu)  // global_addr[56:32]
          | (2u << 30);                         // type = 2 ("image")
  i32x8 g1;
  g1[0] = (int)((2u << 16)      // data_size = 2 -> 4-byte units
                | (1u << 20)    // pad_enable
                | (3u << 22)    // pad_interval code 3 -> every 16 dwords (64 B)
                | (3u << 25));  // pad_amount  code 3 -> 4 dwords (16 B)
  g1[1] = (int)((TD0 & 0xFFFFu) << 16);                  // tensor_dim0[15:0]
  g1[2] = (int)((TD0 >> 16) | ((TD1 & 0xFFFFu) << 16));  // td0 hi | td1 lo
  g1[3] = (int)((TD1 >> 16) | (16u << 16));              // td1 hi | tile_dim0=16
  g1[4] = 128;                                           // tile_dim1 = 128 rows
  g1[5] = 384;                                           // dim0 stride (dwords)
  g1[6] = 0;
  g1[7] = 0;
  i32x4 z4 = {0, 0, 0, 0};
#if __clang_major__ >= 23
  i32x8 z8 = {0, 0, 0, 0, 0, 0, 0, 0};
  __builtin_amdgcn_tensor_load_to_lds(g0, g1, z4, z4, z8, 0);
#else
  __builtin_amdgcn_tensor_load_to_lds(g0, g1, z4, z4, 0);
#endif
}
#endif  // HAVE_TDM

// ---------------------------------------------------------------------------
// 1) fp32 -> fp16 conversion (x and weights)
// ---------------------------------------------------------------------------
__global__ __launch_bounds__(256) void f32_to_f16_kernel(
    const float* __restrict__ src, half_t* __restrict__ dst, int n) {
  int i = (blockIdx.x * 256 + threadIdx.x) * 4;
  if (i < n) {
    float4 v = *(const float4*)(src + i);
    dst[i + 0] = (half_t)v.x;
    dst[i + 1] = (half_t)v.y;
    dst[i + 2] = (half_t)v.z;
    dst[i + 3] = (half_t)v.w;
  }
}

// ---------------------------------------------------------------------------
// 2) QKV projection GEMM: C[t,o] = sum_i X[t,i] * W[o,i] + bias[o]
//    BM=BN=128, BK=32, 256 threads = 8 waves (2x4), wave tile 64x32.
//    TDM feeds LDS (double-buffered); WMMA consumes.
//    mode 0 -> qt[b][h][e][n] (transposed), 1 -> kt, 2 -> vp[b][h][n][64]
// ---------------------------------------------------------------------------
__global__ __launch_bounds__(256) void qkv_gemm_kernel(
    const half_t* __restrict__ xb,
    const half_t* __restrict__ wqh, const half_t* __restrict__ wkh,
    const half_t* __restrict__ wvh,
    const float* __restrict__ bq, const float* __restrict__ bk,
    const float* __restrict__ bv,
    half_t* __restrict__ qt, half_t* __restrict__ kt, half_t* __restrict__ vp) {
  __shared__ half_t As[2][128 * LDSTRIDE];
  __shared__ half_t Bs[2][128 * LDSTRIDE];

  const int mode = blockIdx.z;
  const half_t* w = (mode == 0) ? wqh : ((mode == 1) ? wkh : wvh);
  const float* bias = (mode == 0) ? bq : ((mode == 1) ? bk : bv);

  const int m0 = blockIdx.x * 128;
  const int n0 = blockIdx.y * 128;
  const int tid = threadIdx.x;
  const int lane = tid & 31;
  const int wid = tid >> 5;
  const int wrow = wid >> 2;  // 0..1 -> 64-row half
  const int wcol = wid & 3;   // 0..3 -> 32-col quarter
  const int lrow = tid >> 2;      // 0..63 (prefetch / fallback staging)
  const int lkc = (tid & 3) * 8;  // 0,8,16,24

  const unsigned ldsA = (unsigned)(size_t)&As[0][0];
  const unsigned ldsB = (unsigned)(size_t)&Bs[0][0];
  const unsigned bufBytes = 128 * LDSTRIDE * 2;  // 10240

  auto stage_tile = [&](int k0, int bsel) {
#if HAVE_TDM
    if (wid == 0) {
      tdm_load_tile(xb + (size_t)m0 * kD + k0, ldsA + (unsigned)bsel * bufBytes);
      tdm_load_tile(w + (size_t)n0 * kD + k0, ldsB + (unsigned)bsel * bufBytes);
    }
#else
    uint4 a0 = *(const uint4*)(xb + (size_t)(m0 + lrow) * kD + k0 + lkc);
    uint4 a1 = *(const uint4*)(xb + (size_t)(m0 + lrow + 64) * kD + k0 + lkc);
    uint4 b0 = *(const uint4*)(w + (size_t)(n0 + lrow) * kD + k0 + lkc);
    uint4 b1 = *(const uint4*)(w + (size_t)(n0 + lrow + 64) * kD + k0 + lkc);
    *(uint4*)&As[bsel][lrow * LDSTRIDE + lkc] = a0;
    *(uint4*)&As[bsel][(lrow + 64) * LDSTRIDE + lkc] = a1;
    *(uint4*)&Bs[bsel][lrow * LDSTRIDE + lkc] = b0;
    *(uint4*)&Bs[bsel][(lrow + 64) * LDSTRIDE + lkc] = b1;
#endif
  };
  auto wait_tiles = [&]() {
#if HAVE_TDM
    if (wid == 0) __builtin_amdgcn_s_wait_tensorcnt(0);
#endif
    __syncthreads();
  };

  stage_tile(0, 0);
  wait_tiles();

  v8f acc[4][2] = {};
  int buf = 0;
  const int KT = kD / 32;  // 24
  for (int ktile = 0; ktile < KT; ++ktile) {
    if (ktile < KT - 1) {
      stage_tile((ktile + 1) * 32, buf ^ 1);
      if (ktile < KT - 2) {  // L2 prefetch for the tile after next
        __builtin_prefetch(xb + (size_t)(m0 + lrow) * kD + (ktile + 2) * 32, 0, 1);
        __builtin_prefetch(w + (size_t)(n0 + lrow) * kD + (ktile + 2) * 32, 0, 1);
      }
    }
    const half_t* Ab = &As[buf][(size_t)wrow * 64 * LDSTRIDE];
    const half_t* Bb = &Bs[buf][(size_t)wcol * 32 * LDSTRIDE];
    v16h bf0 = fragB(Bb, LDSTRIDE, lane);
    v16h bf1 = fragB(Bb + 16 * LDSTRIDE, LDSTRIDE, lane);
#pragma unroll
    for (int i = 0; i < 4; ++i) {
      v16h af = fragA(Ab + i * 16 * LDSTRIDE, LDSTRIDE, lane);
      acc[i][0] = wmma16(af, bf0, acc[i][0]);
      acc[i][1] = wmma16(af, bf1, acc[i][1]);
    }
    if (ktile < KT - 1) {
      wait_tiles();
      buf ^= 1;
    }
  }

  // Epilogue: VGPR r holds M=r (lanes 0-15) / M=r+8 (lanes 16-31), N=lane&15.
  const int laneM = (lane < 16) ? 0 : 8;
  const int laneN = lane & 15;
  const int mbase = m0 + wrow * 64;
  const int nbase = n0 + wcol * 32;
#pragma unroll
  for (int i = 0; i < 4; ++i) {
#pragma unroll
    for (int j = 0; j < 2; ++j) {
#pragma unroll
      for (int r = 0; r < 8; ++r) {
        int t = mbase + i * 16 + r + laneM;   // token
        int o = nbase + j * 16 + laneN;       // output channel
        float val = acc[i][j][r] + bias[o];
        int b = t >> 12, n = t & 4095;
        int h = o / kE, e = o - h * kE;
        if (mode == 2) {
          vp[(((size_t)(b * kH + h) * kN) + n) * 64 + e] = (half_t)val;
        } else {
          half_t* dst = (mode == 0) ? qt : kt;
          dst[(((size_t)(b * kH + h) * kE) + e) * kN + n] = (half_t)val;
        }
      }
    }
  }
}

// ---------------------------------------------------------------------------
// 3) Per-(b,h,e) inverse L2 norm over the sequence axis (rows of qt/kt).
// ---------------------------------------------------------------------------
__global__ __launch_bounds__(256) void colnorm_kernel(
    const half_t* __restrict__ qt, const half_t* __restrict__ kt,
    float* __restrict__ rnq, float* __restrict__ rnk) {
  __shared__ float red[256];
  const half_t* src = blockIdx.y ? kt : qt;
  float* dst = blockIdx.y ? rnk : rnq;
  const half_t* p = src + (size_t)blockIdx.x * kN;
  float s = 0.f;
  for (int i = threadIdx.x; i < kN; i += 256) {
    float v = (float)p[i];
    s += v * v;
  }
  red[threadIdx.x] = s;
  __syncthreads();
  for (int off = 128; off > 0; off >>= 1) {
    if (threadIdx.x < off) red[threadIdx.x] += red[threadIdx.x + off];
    __syncthreads();
  }
  if (threadIdx.x == 0) {
    float nrm = sqrtf(red[0]);
    dst[blockIdx.x] = 1.0f / fmaxf(nrm, 1e-12f);
  }
}

// ---------------------------------------------------------------------------
// 4) energy[e,q] = sum_n Q[e,n]*K[q,n]; scale by inv norms; softmax over q;
//    write attnT[q][64] (e-major, /tau, zero-padded e=48..63).
//    One block per (b,h); 8 waves split K=4096 -> ds_add_f32 reduce in LDS.
// ---------------------------------------------------------------------------
__global__ __launch_bounds__(256) void energy_softmax_kernel(
    const half_t* __restrict__ qt, const half_t* __restrict__ kt,
    const float* __restrict__ rnq, const float* __restrict__ rnk,
    const float* __restrict__ tau, half_t* __restrict__ attnT) {
  __shared__ float C[kE * kE];
  const int bh = blockIdx.x;
  const int tid = threadIdx.x;
  const int lane = tid & 31;
  const int wid = tid >> 5;

  for (int i = tid; i < kE * kE; i += 256) C[i] = 0.f;
  __syncthreads();

  const half_t* Q = qt + (size_t)bh * kE * kN;
  const half_t* K = kt + (size_t)bh * kE * kN;

  v8f acc[3][3] = {};
  const int kstart = wid * (kN / 8);  // 512 per wave
  for (int ks = 0; ks < kN / 8; ks += 32) {
    const int k0 = kstart + ks;
    v16h bf[3];
#pragma unroll
    for (int j = 0; j < 3; ++j) bf[j] = fragB(K + (size_t)j * 16 * kN + k0, kN, lane);
#pragma unroll
    for (int i = 0; i < 3; ++i) {
      v16h af = fragA(Q + (size_t)i * 16 * kN + k0, kN, lane);
#pragma unroll
      for (int j = 0; j < 3; ++j) acc[i][j] = wmma16(af, bf[j], acc[i][j]);
    }
  }

  const int laneM = (lane < 16) ? 0 : 8;
  const int laneN = lane & 15;
#pragma unroll
  for (int i = 0; i < 3; ++i)
#pragma unroll
    for (int j = 0; j < 3; ++j)
#pragma unroll
      for (int r = 0; r < 8; ++r)
        atomicAdd(&C[(i * 16 + r + laneM) * kE + j * 16 + laneN], acc[i][j][r]);
  __syncthreads();

  // scale by inverse column norms (exact equivalent of normalizing q,k)
  for (int idx = tid; idx < kE * kE; idx += 256) {
    int e = idx / kE, q = idx - e * kE;
    C[idx] *= rnq[bh * kE + e] * rnk[bh * kE + q];
  }
  __syncthreads();

  if (tid < kE) {
    const int e = tid;
    float mx = -3.0e38f;
    for (int q = 0; q < kE; ++q) mx = fmaxf(mx, C[e * kE + q]);
    float s = 0.f;
    for (int q = 0; q < kE; ++q) {
      float ex = __expf(C[e * kE + q] - mx);
      C[e * kE + q] = ex;
      s += ex;
    }
    float inv = 1.0f / s;
    for (int q = 0; q < kE; ++q) C[e * kE + q] *= inv;
  }
  __syncthreads();

  const float it = 1.0f / tau[0];
  half_t* out = attnT + (size_t)bh * kE * 64;
  for (int idx = tid; idx < kE * 64; idx += 256) {
    int q = idx >> 6, e = idx & 63;
    float v = (e < kE) ? C[e * kE + q] * it : 0.f;
    out[q * 64 + e] = (half_t)v;
  }
}

// ---------------------------------------------------------------------------
// 5) out_pre[n, q] = sum_e V[n,e] * attnT[q,e]   (K padded to 64; pad rows of
//    attnT are zero so V's pad columns are don't-care). Writes yb[t][h*48+q].
// ---------------------------------------------------------------------------
__global__ __launch_bounds__(256) void av_gemm_kernel(
    const half_t* __restrict__ vp, const half_t* __restrict__ attnT,
    half_t* __restrict__ yb) {
  const int bh = blockIdx.y;
  const int b = bh >> 4, h = bh & 15;
  const int tid = threadIdx.x;
  const int lane = tid & 31;
  const int wid = tid >> 5;
  const int n0 = blockIdx.x * 128 + wid * 16;

  const half_t* V = vp + (size_t)bh * kN * 64;
  const half_t* A = attnT + (size_t)bh * kE * 64;

  v8f acc[3] = {};
#pragma unroll
  for (int ks = 0; ks < 64; ks += 32) {
    v16h af = fragA(V + (size_t)n0 * 64 + ks, 64, lane);
#pragma unroll
    for (int j = 0; j < 3; ++j) {
      v16h bf = fragB(A + (size_t)j * 16 * 64 + ks, 64, lane);
      acc[j] = wmma16(af, bf, acc[j]);
    }
  }

  const int laneM = (lane < 16) ? 0 : 8;
  const int laneN = lane & 15;
#pragma unroll
  for (int j = 0; j < 3; ++j)
#pragma unroll
    for (int r = 0; r < 8; ++r) {
      int n = n0 + r + laneM;
      int q = j * 16 + laneN;
      yb[(size_t)(b * kN + n) * kD + h * kE + q] = (half_t)acc[j][r];
    }
}

// ---------------------------------------------------------------------------
// 6) Output projection: out[t,o] = sum_c yb[t,c]*Wo[o,c] + bo[o], fp32 out.
//    Same TDM-fed WMMA pipeline as the QKV GEMM.
// ---------------------------------------------------------------------------
__global__ __launch_bounds__(256) void out_proj_kernel(
    const half_t* __restrict__ yb, const half_t* __restrict__ woh,
    const float* __restrict__ bo, float* __restrict__ out) {
  __shared__ half_t As[2][128 * LDSTRIDE];
  __shared__ half_t Bs[2][128 * LDSTRIDE];

  const int m0 = blockIdx.x * 128;
  const int n0 = blockIdx.y * 128;
  const int tid = threadIdx.x;
  const int lane = tid & 31;
  const int wid = tid >> 5;
  const int wrow = wid >> 2;
  const int wcol = wid & 3;
  const int lrow = tid >> 2;
  const int lkc = (tid & 3) * 8;

  const unsigned ldsA = (unsigned)(size_t)&As[0][0];
  const unsigned ldsB = (unsigned)(size_t)&Bs[0][0];
  const unsigned bufBytes = 128 * LDSTRIDE * 2;

  auto stage_tile = [&](int k0, int bsel) {
#if HAVE_TDM
    if (wid == 0) {
      tdm_load_tile(yb + (size_t)m0 * kD + k0, ldsA + (unsigned)bsel * bufBytes);
      tdm_load_tile(woh + (size_t)n0 * kD + k0, ldsB + (unsigned)bsel * bufBytes);
    }
#else
    uint4 a0 = *(const uint4*)(yb + (size_t)(m0 + lrow) * kD + k0 + lkc);
    uint4 a1 = *(const uint4*)(yb + (size_t)(m0 + lrow + 64) * kD + k0 + lkc);
    uint4 b0 = *(const uint4*)(woh + (size_t)(n0 + lrow) * kD + k0 + lkc);
    uint4 b1 = *(const uint4*)(woh + (size_t)(n0 + lrow + 64) * kD + k0 + lkc);
    *(uint4*)&As[bsel][lrow * LDSTRIDE + lkc] = a0;
    *(uint4*)&As[bsel][(lrow + 64) * LDSTRIDE + lkc] = a1;
    *(uint4*)&Bs[bsel][lrow * LDSTRIDE + lkc] = b0;
    *(uint4*)&Bs[bsel][(lrow + 64) * LDSTRIDE + lkc] = b1;
#endif
  };
  auto wait_tiles = [&]() {
#if HAVE_TDM
    if (wid == 0) __builtin_amdgcn_s_wait_tensorcnt(0);
#endif
    __syncthreads();
  };

  stage_tile(0, 0);
  wait_tiles();

  v8f acc[4][2] = {};
  int buf = 0;
  const int KT = kD / 32;
  for (int ktile = 0; ktile < KT; ++ktile) {
    if (ktile < KT - 1) {
      stage_tile((ktile + 1) * 32, buf ^ 1);
      if (ktile < KT - 2) {
        __builtin_prefetch(yb + (size_t)(m0 + lrow) * kD + (ktile + 2) * 32, 0, 1);
        __builtin_prefetch(woh + (size_t)(n0 + lrow) * kD + (ktile + 2) * 32, 0, 1);
      }
    }
    const half_t* Ab = &As[buf][(size_t)wrow * 64 * LDSTRIDE];
    const half_t* Bb = &Bs[buf][(size_t)wcol * 32 * LDSTRIDE];
    v16h bf0 = fragB(Bb, LDSTRIDE, lane);
    v16h bf1 = fragB(Bb + 16 * LDSTRIDE, LDSTRIDE, lane);
#pragma unroll
    for (int i = 0; i < 4; ++i) {
      v16h af = fragA(Ab + i * 16 * LDSTRIDE, LDSTRIDE, lane);
      acc[i][0] = wmma16(af, bf0, acc[i][0]);
      acc[i][1] = wmma16(af, bf1, acc[i][1]);
    }
    if (ktile < KT - 1) {
      wait_tiles();
      buf ^= 1;
    }
  }

  const int laneM = (lane < 16) ? 0 : 8;
  const int laneN = lane & 15;
  const int mbase = m0 + wrow * 64;
  const int nbase = n0 + wcol * 32;
#pragma unroll
  for (int i = 0; i < 4; ++i)
#pragma unroll
    for (int j = 0; j < 2; ++j)
#pragma unroll
      for (int r = 0; r < 8; ++r) {
        int t = mbase + i * 16 + r + laneM;
        int o = nbase + j * 16 + laneN;
        out[(size_t)t * kD + o] = acc[i][j][r] + bo[o];
      }
}

// ---------------------------------------------------------------------------
// Launch
// ---------------------------------------------------------------------------
extern "C" void kernel_launch(void* const* d_in, const int* in_sizes, int n_in,
                              void* d_out, int out_size, void* d_ws,
                              size_t ws_size, hipStream_t stream) {
  (void)in_sizes; (void)n_in; (void)out_size; (void)ws_size;
  const float* x  = (const float*)d_in[0];
  const float* wq = (const float*)d_in[1];
  const float* bq = (const float*)d_in[2];
  const float* wk = (const float*)d_in[3];
  const float* bk = (const float*)d_in[4];
  const float* wv = (const float*)d_in[5];
  const float* bv = (const float*)d_in[6];
  const float* wo = (const float*)d_in[7];
  const float* bo = (const float*)d_in[8];
  const float* tau = (const float*)d_in[9];

  char* ws = (char*)d_ws;
  size_t off = 0;
  auto alloc = [&](size_t bytes) -> void* {
    void* p = ws + off;
    off = (off + bytes + 255) & ~(size_t)255;
    return p;
  };

  const size_t NX = (size_t)kTOK * kD;   // 50,331,648
  const size_t NW = (size_t)kD * kD;     // 589,824

  half_t* xb   = (half_t*)alloc(NX * 2);  // also reused as yb after QKV
  half_t* wqh  = (half_t*)alloc(NW * 2);
  half_t* wkh  = (half_t*)alloc(NW * 2);
  half_t* wvh  = (half_t*)alloc(NW * 2);
  half_t* woh  = (half_t*)alloc(NW * 2);
  half_t* qt   = (half_t*)alloc(NX * 2);                       // [B][H][E][N]
  half_t* kt   = (half_t*)alloc(NX * 2);                       // [B][H][E][N]
  half_t* vp   = (half_t*)alloc((size_t)kB * kH * kN * 64 * 2);// [B][H][N][64]
  float*  rnq  = (float*)alloc((size_t)kB * kH * kE * 4);
  float*  rnk  = (float*)alloc((size_t)kB * kH * kE * 4);
  half_t* attnT = (half_t*)alloc((size_t)kB * kH * kE * 64 * 2);
  half_t* yb = xb;  // safe alias: x no longer needed after QKV GEMM

  // 1) fp32 -> fp16
  f32_to_f16_kernel<<<(int)(NX / 1024), 256, 0, stream>>>(x, xb, (int)NX);
  f32_to_f16_kernel<<<(int)(NW / 1024), 256, 0, stream>>>(wq, wqh, (int)NW);
  f32_to_f16_kernel<<<(int)(NW / 1024), 256, 0, stream>>>(wk, wkh, (int)NW);
  f32_to_f16_kernel<<<(int)(NW / 1024), 256, 0, stream>>>(wv, wvh, (int)NW);
  f32_to_f16_kernel<<<(int)(NW / 1024), 256, 0, stream>>>(wo, woh, (int)NW);

  // 2) QKV projections (TDM -> LDS -> WMMA)
  qkv_gemm_kernel<<<dim3(kTOK / 128, kD / 128, 3), 256, 0, stream>>>(
      xb, wqh, wkh, wvh, bq, bk, bv, qt, kt, vp);

  // 3) inverse sequence-axis L2 norms
  colnorm_kernel<<<dim3(kB * kH * kE, 2), 256, 0, stream>>>(qt, kt, rnq, rnk);

  // 4) energy + softmax -> attnT
  energy_softmax_kernel<<<kB * kH, 256, 0, stream>>>(qt, kt, rnq, rnk, tau,
                                                     attnT);

  // 5) attention-weighted V -> yb (pre-projection), channel-major
  av_gemm_kernel<<<dim3(kN / 128, kB * kH), 256, 0, stream>>>(vp, attnT, yb);

  // 6) output projection + bias -> fp32 d_out
  out_proj_kernel<<<dim3(kTOK / 128, kD / 128), 256, 0, stream>>>(
      yb, woh, bo, (float*)d_out);
}
